// TinyNeRF_28492813042019
// MI455X (gfx1250) — matrix operand order repaired and checked by reading, not compiled, not run on hardware
//
#include <hip/hip_runtime.h>
#include <hip/hip_bf16.h>
#include <math.h>

// ---------------------------------------------------------------------------
// TinyNeRF forward on MI455X (gfx1250, wave32) via V_WMMA_F32_16X16X32_F16.
//
// One wave owns 16 rays and marches 64 samples. MLP evaluated as transposed
// GEMMs out^T = W^T @ act^T; the WMMA D fragment (lane = point, reg =
// out-feature) restages into the next layer's B fragment via ds_store_b128 /
// ds_load_b128 in a per-wave LDS row.  Biases are folded into the weights
// through a constant-1 input feature (feature 83), so C = inline 0.
//
// All 83 weight A-fragments (664 VGPRs) are preloaded into registers before
// the sample loop — CDNA5 wave32 allows 1024 VGPRs/wave — so the 64-step loop
// issues no global loads at all.  Leaky-ReLU is applied on packed f16 pairs
// (v_pk_mul_f16 + v_pk_max_num_f16) after the f32->f16 pack.
// ---------------------------------------------------------------------------

typedef _Float16 v16h  __attribute__((ext_vector_type(16)));
typedef _Float16 half8 __attribute__((ext_vector_type(8)));
typedef float    v8f   __attribute__((ext_vector_type(8)));

#define NERF_STEPS  64
#define NERF_HIDDEN 80
#define NERF_NSKIP  5
#define NFRAG_TOTAL 83   // 5 (W0) + 5*15 (Wh) + 3 (Wout)

// A-fragment (16-bit A 16x32) k pattern: element e in lane-half h holds
// k = e + 16h (contiguous, matching the B fragments' LDS reads)
__device__ __forceinline__ int a_kpat(int e, int h) {
    return e + 16 * h;
}

// ---------------------------------------------------------------------------
// Prep kernel: transpose/pad weights into WMMA A-fragment lane layout (f16),
// bias placed at input-row k==83 (k==3 for the W0 tile, whose k-tile is the
// pts region).  Staged activations carry 1.0 at feature 83, so C can be zero.
//   frags 0..4   : W0^T  tile f; k 0..2 = W0, k 3 = b0
//   frags 5..79  : Wh^T  layer i, out-tile f, k-tile t (idx = 5 + 15i + 3f + t)
//                  kabs 0..82 = Wh, kabs 83 = bh[i]
//   frags 80..82 : Wout^T (channels 0..3); kabs 0..79 = Wout, kabs 83 = bout
// ---------------------------------------------------------------------------
__global__ void nerf_prep_wfrags(const float* __restrict__ W0,
                                 const float* __restrict__ b0,
                                 const float* __restrict__ Wh,
                                 const float* __restrict__ bh,
                                 const float* __restrict__ Wout,
                                 const float* __restrict__ bout,
                                 _Float16* __restrict__ wsA) {
    int tid  = blockIdx.x * blockDim.x + threadIdx.x;
    int frag = tid >> 5;
    if (frag >= NFRAG_TOTAL) return;
    int lane = tid & 31;
    int n = lane & 15, h = lane >> 4;

    _Float16 vals[16];
#pragma unroll
    for (int e = 0; e < 16; ++e) {
        int k = a_kpat(e, h);
        float v = 0.0f;
        if (frag < 5) {                         // W0: (3,80) [in][out]
            int m = 16 * frag + n;
            if (k < 3)       v = W0[k * NERF_HIDDEN + m];
            else if (k == 3) v = b0[m];         // bias row (feature 83)
        } else if (frag < 80) {                 // Wh: (5,83,80), bh: (5,80)
            int q = frag - 5;
            int i = q / 15, r2 = q % 15;
            int f = r2 / 3, t = r2 % 3;
            int m    = 16 * f + n;
            int kabs = 32 * t + k;
            if (kabs < NERF_HIDDEN + 3)
                v = Wh[(i * (NERF_HIDDEN + 3) + kabs) * NERF_HIDDEN + m];
            else if (kabs == 83)
                v = bh[i * NERF_HIDDEN + m];    // bias row
        } else {                                // Wout: (80,4), bout: (4,)
            int t    = frag - 80;
            int m    = n;
            int kabs = 32 * t + k;
            if (m < 4) {
                if (kabs < NERF_HIDDEN) v = Wout[kabs * 4 + m];
                else if (kabs == 83)    v = bout[m];  // bias row
            }
        }
        vals[e] = (_Float16)v;
    }
    _Float16* dst = wsA + frag * 512 + lane * 16;
    *(half8*)(dst)     = *(half8*)&vals[0];
    *(half8*)(dst + 8) = *(half8*)&vals[8];
}

__device__ __forceinline__ float fat_sigmoid(float v) {
    return (1.0f / (1.0f + __expf(-v))) * 1.002f - 0.001f;
}

// leaky_relu on packed f16: pack first (4x v_cvt_pk_f16_f32), then
// v_pk_mul_f16 (splat 0.01) + v_pk_max_num_f16 on pairs.
__device__ __forceinline__ half8 act_cvt(v8f v) {
    half8 hv = __builtin_convertvector(v, half8);
    half8 hs = hv * (_Float16)0.01f;          // scalar splat -> v_pk_mul_f16
    return __builtin_elementwise_max(hv, hs); // -> v_pk_max_num_f16
}

// ---------------------------------------------------------------------------
// Main kernel: ONE wave per block (max VGPR budget), one wave = 16 rays,
// loops 64 samples.  Per-wave LDS row (per point, _Float16[128]):
//   [0..79] hidden   [80..82] pts   [83] 1.0 (bias input)   [84..111] zeros
// ---------------------------------------------------------------------------
__global__ __launch_bounds__(32) void nerf_fwd(
        const float* __restrict__ rays,
        const _Float16* __restrict__ wsA,
        float* __restrict__ out) {
    __shared__ _Float16 smem[16][128];

    const int lane = threadIdx.x & 31;
    const int n = lane & 15, h = lane >> 4;
    const int ray = blockIdx.x * 16 + n;

    const float* rp = rays + ray * 6;
    float rox = rp[0], roy = rp[1], roz = rp[2];
    float rdx = rp[3], rdy = rp[4], rdz = rp[5];
    float rnorm = sqrtf(rdx * rdx + rdy * rdy + rdz * rdz);

    const v16h* __restrict__ wA = (const v16h*)wsA;
    _Float16* arow = &smem[n][0];

    // ---- Preload ALL weight fragments into VGPRs (83 * 8 = 664 VGPRs) ------
    v16h w[NFRAG_TOTAL];
#pragma unroll
    for (int q = 0; q < NFRAG_TOTAL; ++q) w[q] = wA[q * 32 + lane];

    v8f czero;
#pragma unroll
    for (int r = 0; r < 8; ++r) czero[r] = 0.0f;

    const _Float16 hzero = (_Float16)0.0f;
    float T = 1.0f, acc0 = 0.0f, acc1 = 0.0f, acc2 = 0.0f;

    for (int s = 0; s < NERF_STEPS; ++s) {
        float ts = (float)s * (1.0f / 63.0f);

        // Stage [pts | 1 | 0...] at features [80 + 16h .. 95 + 16h].
        // Branchless (v_cndmask) to avoid EXEC save/restore around the store.
        half8 p0 = {}, p1 = {};
        p0[0] = h ? hzero : (_Float16)(rox + ts * rdx);
        p0[1] = h ? hzero : (_Float16)(roy + ts * rdy);
        p0[2] = h ? hzero : (_Float16)(roz + ts * rdz);
        p0[3] = h ? hzero : (_Float16)1.0f;     // bias input (feature 83)
        *(half8*)(arow + 80 + 16 * h) = p0;
        *(half8*)(arow + 88 + 16 * h) = p1;
        asm volatile("" ::: "memory");  // keep DS program order; HW DS is in-order per wave

        // ---- layer 0: h^T = leaky_relu(W0aug^T @ [pts;1]^T), 5 indep WMMAs --
        v16h bf_in = *(const v16h*)(arow + 80 + 16 * h);
        v8f hacc[5];
#pragma unroll
        for (int f = 0; f < 5; ++f)
            hacc[f] = __builtin_amdgcn_wmma_f32_16x16x32_f16(
                          false, w[f], false, bf_in, (short)0, czero, false, false);
#pragma unroll
        for (int f = 0; f < 5; ++f)
            *(half8*)(arow + 16 * f + 8 * h) = act_cvt(hacc[f]);

        // ---- skip layers: t-outer / f-inner => 5 independent WMMA chains ----
#pragma unroll
        for (int i = 0; i < NERF_NSKIP; ++i) {
            asm volatile("" ::: "memory");
            v16h bf0 = *(const v16h*)(arow +  0 + 16 * h);
            v16h bf1 = *(const v16h*)(arow + 32 + 16 * h);
            v16h bf2 = *(const v16h*)(arow + 64 + 16 * h);  // covers [h|pts|1|0]
            v8f acc[5];
#pragma unroll
            for (int f = 0; f < 5; ++f)
                acc[f] = __builtin_amdgcn_wmma_f32_16x16x32_f16(
                             false, w[5 + 15 * i + 3 * f + 0], false, bf0,
                             (short)0, czero, false, false);
#pragma unroll
            for (int f = 0; f < 5; ++f)
                acc[f] = __builtin_amdgcn_wmma_f32_16x16x32_f16(
                             false, w[5 + 15 * i + 3 * f + 1], false, bf1,
                             (short)0, acc[f], false, false);
#pragma unroll
            for (int f = 0; f < 5; ++f)
                acc[f] = __builtin_amdgcn_wmma_f32_16x16x32_f16(
                             false, w[5 + 15 * i + 3 * f + 2], false, bf2,
                             (short)0, acc[f], false, false);
            asm volatile("" ::: "memory");
#pragma unroll
            for (int f = 0; f < 5; ++f)
                *(half8*)(arow + 16 * f + 8 * h) = act_cvt(acc[f]);
        }

        // ---- output layer: raw^T[4x16] = Woutaug^T @ [h; pts; 1]^T ----------
        asm volatile("" ::: "memory");
        v16h of0 = *(const v16h*)(arow +  0 + 16 * h);
        v16h of1 = *(const v16h*)(arow + 32 + 16 * h);
        v16h of2 = *(const v16h*)(arow + 64 + 16 * h);
        v8f oc = __builtin_amdgcn_wmma_f32_16x16x32_f16(
                     false, w[80], false, of0, (short)0, czero, false, false);
        oc = __builtin_amdgcn_wmma_f32_16x16x32_f16(
                     false, w[81], false, of1, (short)0, oc, false, false);
        oc = __builtin_amdgcn_wmma_f32_16x16x32_f16(
                     false, w[82], false, of2, (short)0, oc, false, false);

        // ---- volume rendering recurrence (valid on half-0 lanes) ------------
        float d0 = oc[0], g1 = oc[1], g2 = oc[2], g3 = oc[3];
        float xs = d0 - 1.0f;
        float sigma = (xs > 20.0f) ? xs : log1pf(__expf(xs));   // softplus
        float dist  = ((s == NERF_STEPS - 1) ? 1e10f : (1.0f / 63.0f)) * rnorm;
        float alpha = 1.0f - __expf(-sigma * dist);
        float wgt   = alpha * T;
        T = T * (1.0f - alpha + 1e-10f);
        acc0 += wgt * fat_sigmoid(g1);
        acc1 += wgt * fat_sigmoid(g2);
        acc2 += wgt * fat_sigmoid(g3);
    }

    if (h == 0) {
        float* op = out + ray * 3;
        op[0] = acc0; op[1] = acc1; op[2] = acc2;
    }
}

// ---------------------------------------------------------------------------
extern "C" void kernel_launch(void* const* d_in, const int* in_sizes, int n_in,
                              void* d_out, int out_size, void* d_ws, size_t ws_size,
                              hipStream_t stream) {
    const float* rays = (const float*)d_in[0];
    const float* W0   = (const float*)d_in[1];
    const float* b0   = (const float*)d_in[2];
    const float* Wh   = (const float*)d_in[3];
    const float* bh   = (const float*)d_in[4];
    const float* Wout = (const float*)d_in[5];
    const float* bout = (const float*)d_in[6];
    float* out = (float*)d_out;
    _Float16* wsA = (_Float16*)d_ws;   // needs 83 * 1024 = 84,992 bytes

    nerf_prep_wfrags<<<(NFRAG_TOTAL * 32 + 127) / 128, 128, 0, stream>>>(
        W0, b0, Wh, bh, Wout, bout, wsA);

    int nrays  = in_sizes[0] / 6;      // 16384
    int nwaves = nrays / 16;           // 1024 single-wave blocks
    nerf_fwd<<<nwaves, 32, 0, stream>>>(rays, wsA, out);
}
